// LISTA_37976100831401
// MI455X (gfx1250) — matrix operand order repaired and checked by reading, not compiled, not run on hardware
//
#include <hip/hip_runtime.h>

// ---------- CDNA5 WMMA types ----------
typedef __attribute__((ext_vector_type(16))) __bf16 v16bf;
typedef __attribute__((ext_vector_type(8)))  float  v8f;

// Problem constants (fixed by the reference)
constexpr int BATCH = 16384;
constexpr int NIN   = 256;    // n
constexpr int MDIM  = 1024;   // m
constexpr int TSTEP = 16;     // LISTA iterations

// Tiling
constexpr int BM  = 32;           // batch rows per workgroup
constexpr int LDZ = MDIM + 8;     // bf16 row stride for Z in LDS (bank-conflict pad)
constexpr int LDB = MDIM + 4;     // f32 row stride for Bmat in LDS
constexpr int LDX = NIN + 8;      // bf16 row stride for X tile in LDS

union FragBF { v16bf v; uint4 u[2]; };

// A matrix 16x32 bf16: lane = 16*g + m; VGPR0-3 hold K = 16g..16g+7,
// VGPR4-7 hold K = 16g+8..16g+15  -> two 16B loads per lane.
__device__ __forceinline__ v16bf load_a_frag(const __bf16* row, int g) {
  FragBF f;
  f.u[0] = *reinterpret_cast<const uint4*>(row + 16 * g);
  f.u[1] = *reinterpret_cast<const uint4*>(row + 16 * g + 8);
  return f.v;
}

// B matrix 32x16 bf16: lane = 16*g + n; lanes 0-15 hold K=0..15, lanes 16-31 hold
// K=16..31, 2 consecutive K per VGPR -> one contiguous 32B span per lane.
__device__ __forceinline__ v16bf load_b_frag(const __bf16* row, int g) {
  FragBF f;
  const uint4* q = reinterpret_cast<const uint4*>(row + 16 * g);
  f.u[0] = q[0];
  f.u[1] = q[1];
  return f.v;
}

__device__ __forceinline__ float soft_thr(float x, float th) {
  float a = fmaxf(fabsf(x) - th, 0.0f);
  return (x < 0.0f) ? -a : a;
}

__device__ __forceinline__ v8f wmma_bf16(v16bf a, v16bf b, v8f c) {
  return __builtin_amdgcn_wmma_f32_16x16x32_bf16(false, a, false, b,
                                                 (short)0, c, false, false);
}

// ---------- one-time f32 -> bf16 conversion of S and We into workspace ----------
__global__ void lista_cvt_kernel(const float* __restrict__ S,
                                 const float* __restrict__ We,
                                 __bf16* __restrict__ Sb,
                                 __bf16* __restrict__ Web) {
  int idx = blockIdx.x * blockDim.x + threadIdx.x;
  const int nS = MDIM * MDIM;
  const int nW = MDIM * NIN;
  if (idx < nS)            Sb[idx]       = (__bf16)S[idx];
  else if (idx < nS + nW)  Web[idx - nS] = (__bf16)We[idx - nS];
}

// ---------- main persistent-tile LISTA kernel ----------
// 256 threads = 8 waves. Each wave owns a 128-column range (8 N-tiles) and BOTH
// 16-row M-tiles of the workgroup's 32 batch rows (register reuse of S frags).
__global__ __launch_bounds__(256, 1)
void lista_kernel(const float* __restrict__ X,
                  const float* __restrict__ theta,
                  const __bf16* __restrict__ Web,
                  const __bf16* __restrict__ Sb,
                  float* __restrict__ Out) {
  __shared__ float  Bsm[BM * LDB];        // 131584 B : B = X@We^T, f32
  __shared__ __bf16 Zs[2 * BM * LDZ];     // 132096 B : double-buffered Z, bf16
  __shared__ __bf16 Xs[BM * LDX];         //  16896 B : X tile, bf16

  const int tid  = threadIdx.x;
  const int lane = tid & 31;
  const int wid  = tid >> 5;      // 0..7
  const int g    = lane >> 4;     // half-wave select
  const int ln   = lane & 15;
  const int rowg = blockIdx.x * BM;

  // ---- phase A: stage X tile (32x256) into LDS as bf16 ----
  for (int idx = tid; idx < BM * NIN; idx += 256) {
    int r = idx >> 8;             // NIN == 256
    int c = idx & (NIN - 1);
    Xs[r * LDX + c] = (__bf16)X[(size_t)(rowg + r) * NIN + c];
  }
  __syncthreads();

  // per-lane theta for this wave's 8 column tiles (C/D layout: N = lane&15)
  const int jw = wid * 128;
  float th[8];
#pragma unroll
  for (int nt = 0; nt < 8; ++nt) th[nt] = theta[jw + nt * 16 + ln];

  // ---- phase B: Bmat = X @ We^T (K = 256), Z0 = soft(Bmat) ----
#pragma unroll
  for (int grp = 0; grp < 2; ++grp) {
    const int jb = jw + grp * 64;
    v8f acc[2][4] = {};
#pragma unroll 2
    for (int k = 0; k < NIN; k += 32) {
      v16bf a0 = load_a_frag(&Xs[ln * LDX + k], g);
      v16bf a1 = load_a_frag(&Xs[(16 + ln) * LDX + k], g);
#pragma unroll
      for (int t = 0; t < 4; ++t) {
        v16bf b = load_b_frag(&Web[(size_t)(jb + t * 16 + ln) * NIN + k], g);
        acc[0][t] = wmma_bf16(a0, b, acc[0][t]);
        acc[1][t] = wmma_bf16(a1, b, acc[1][t]);
      }
    }
#pragma unroll
    for (int mi = 0; mi < 2; ++mi)
#pragma unroll
      for (int t = 0; t < 4; ++t) {
        const int   col = jb + t * 16 + ln;
        const float thv = th[grp * 4 + t];
#pragma unroll
        for (int r = 0; r < 8; ++r) {
          const int mrow = mi * 16 + g * 8 + r;     // C/D: M = 8*g + r
          const float c  = acc[mi][t][r];
          Bsm[mrow * LDB + col] = c;
          Zs[mrow * LDZ + col]  = (__bf16)soft_thr(c, thv);
        }
      }
  }
  __syncthreads();

  // ---- T LISTA iterations: Z_next = soft(B + Z_cur @ S^T) ----
#pragma unroll 1
  for (int it = 0; it < TSTEP; ++it) {
    const __bf16* Zcur = &Zs[(it & 1) * BM * LDZ];
    __bf16*       Znxt = &Zs[((it + 1) & 1) * BM * LDZ];
    const bool    last = (it == TSTEP - 1);

#pragma unroll
    for (int grp = 0; grp < 2; ++grp) {
      const int jb = jw + grp * 64;
      v8f acc[2][4] = {};
#pragma unroll 2
      for (int k = 0; k < MDIM; k += 32) {
        v16bf a0 = load_a_frag(&Zcur[ln * LDZ + k], g);
        v16bf a1 = load_a_frag(&Zcur[(16 + ln) * LDZ + k], g);
#pragma unroll
        for (int t = 0; t < 4; ++t) {
          v16bf b = load_b_frag(&Sb[(size_t)(jb + t * 16 + ln) * MDIM + k], g);
          acc[0][t] = wmma_bf16(a0, b, acc[0][t]);
          acc[1][t] = wmma_bf16(a1, b, acc[1][t]);
        }
      }
#pragma unroll
      for (int mi = 0; mi < 2; ++mi)
#pragma unroll
        for (int t = 0; t < 4; ++t) {
          const int   col = jb + t * 16 + ln;
          const float thv = th[grp * 4 + t];
#pragma unroll
          for (int r = 0; r < 8; ++r) {
            const int mrow = mi * 16 + g * 8 + r;
            const float c  = acc[mi][t][r] + Bsm[mrow * LDB + col];
            const float z  = soft_thr(c, thv);
            Znxt[mrow * LDZ + col] = (__bf16)z;
            if (last) Out[(size_t)(rowg + mrow) * MDIM + col] = z;
          }
        }
    }
    __syncthreads();
  }
}

// ---------- launch ----------
extern "C" void kernel_launch(void* const* d_in, const int* in_sizes, int n_in,
                              void* d_out, int out_size, void* d_ws, size_t ws_size,
                              hipStream_t stream) {
  (void)in_sizes; (void)n_in; (void)out_size; (void)ws_size;
  const float* X     = (const float*)d_in[0];   // (16384, 256)
  const float* We    = (const float*)d_in[1];   // (1024, 256)
  const float* S     = (const float*)d_in[2];   // (1024, 1024)
  const float* theta = (const float*)d_in[3];   // (1024,)
  float*       Out   = (float*)d_out;           // (16384, 1024)

  __bf16* Sb  = (__bf16*)d_ws;                  // 2 MB
  __bf16* Web = Sb + (size_t)MDIM * MDIM;       // 0.5 MB

  const int ncvt = MDIM * MDIM + MDIM * NIN;
  lista_cvt_kernel<<<(ncvt + 255) / 256, 256, 0, stream>>>(S, We, Sb, Web);
  lista_kernel<<<BATCH / BM, 256, 0, stream>>>(X, theta, Web, Sb, Out);
}
